// Slayer2LayerMLP_53291954209114
// MI455X (gfx1250) — matrix operand order repaired and checked by reading, not compiled, not run on hardware
//
#include <hip/hip_runtime.h>

typedef _Float16 h16;
typedef _Float16 v16h __attribute__((ext_vector_type(16)));
typedef _Float16 v8h  __attribute__((ext_vector_type(8)));
typedef float    v8f  __attribute__((ext_vector_type(8)));
typedef float    v4f  __attribute__((ext_vector_type(4)));
typedef unsigned int u32x4 __attribute__((ext_vector_type(4)));
typedef int          i32x4 __attribute__((ext_vector_type(4)));
typedef int          i32x8 __attribute__((ext_vector_type(8)));

#define B_    32
#define IN_   2048
#define HID_  1024
#define OUT_  512
#define T_    300
#define TP_   304                       // T padded to multiple of 16
#define THETA_     10.0f
#define ALPHA_SR_  0.9048374180359595f  // exp(-1/10)
#define ALPHA_REF_ 0.6065306597126334f  // exp(-1/2)
#define REF_SCALE_ 20.0f                // 2*theta

// ---------------------------------------------------------------------------
// Kernel 1: transpose + convert x[B,IN,T] f32 -> xT[B,Tp,IN] f16 (zero pad t)
// ---------------------------------------------------------------------------
__global__ __launch_bounds__(256)
void transpose_x_kernel(const float* __restrict__ x, h16* __restrict__ xT) {
  __shared__ float tile[32][33];
  const int b  = blockIdx.z;
  const int t0 = blockIdx.x * 32;
  const int i0 = blockIdx.y * 32;
  const int tx = threadIdx.x;   // 0..31
  const int ty = threadIdx.y;   // 0..7
  const float* xb = x + (size_t)b * IN_ * T_;
#pragma unroll
  for (int k = 0; k < 4; ++k) {
    int il = ty + k * 8;
    int t  = t0 + tx;
    float v = (t < T_) ? xb[(size_t)(i0 + il) * T_ + t] : 0.0f;
    tile[il][tx] = v;
  }
  __syncthreads();
  h16* xTb = xT + (size_t)b * TP_ * IN_;
#pragma unroll
  for (int k = 0; k < 4; ++k) {
    int tl = ty + k * 8;
    int t  = t0 + tl;
    if (t < TP_) xTb[(size_t)t * IN_ + i0 + tx] = (h16)tile[tx][tl];
  }
}

// ---------------------------------------------------------------------------
// Kernel 2: pack W[Cout,Cin] f32 into WMMA B-fragment order, f16.
// [ntile][kchunk32][lane 0..31][16 f16]: lane holds column n=ntile*16+(lane&15),
// K = kchunk*32 + (lane>>4)*16 + e.
// ---------------------------------------------------------------------------
__global__ __launch_bounds__(256)
void pack_w_kernel(const float* __restrict__ W, h16* __restrict__ Wp,
                   int Cout, int Cin) {
  int tid = blockIdx.x * blockDim.x + threadIdx.x;
  int total = Cout * Cin;
  if (tid >= total) return;
  int e     = tid & 15;
  int lane  = (tid >> 4) & 31;
  int rest  = tid >> 9;              // ntile*nkc + kc
  int nkc   = Cin >> 5;
  int kc    = rest % nkc;
  int ntile = rest / nkc;
  int n = ntile * 16 + (lane & 15);
  int k = kc * 32 + ((lane >> 4) << 4) + e;
  Wp[tid] = (h16)W[(size_t)n * Cin + k];
}

// ---------------------------------------------------------------------------
// Kernel 3: fused GEMM (WMMA f16->f32) + LIF scan. 128 threads = 4 waves,
// wave owns 32 neurons (two 16x16 accumulators sharing one A fragment).
// X tile staged to LDS via the Tensor Data Mover (TENSORcnt) when available.
// ---------------------------------------------------------------------------
template <bool F16OUT>
__global__ __launch_bounds__(128)
void slayer_layer_kernel(const h16* __restrict__ xT,   // [B, TP, Cin] f16
                         const h16* __restrict__ Wp,   // packed fragments
                         h16* __restrict__ sOut,       // [B, TP, Cout] if F16OUT
                         float* __restrict__ yOut,     // [B, Cout, T ] else
                         int Cin, int Cout) {
  constexpr int KSTAGE = 256;         // f16 per row staged in LDS (512 B)
  constexpr int LROW   = KSTAGE + 8;  // +16B/row pad (4 dwords): conflict relief
  constexpr int KKS    = KSTAGE / 32; // 8 WMMA k-steps per stage
  __shared__ h16   ldsx[16 * LROW];
  __shared__ float zbuf[4][16][33];
  __shared__ h16   sbuf[4][16][40];

  const int tid   = threadIdx.x;
  const int wave  = tid >> 5;
  const int lane  = tid & 31;
  const int b     = blockIdx.x;
  const int n0    = blockIdx.y * 128 + wave * 32;  // 32 neurons per wave
  const int nkc   = Cin >> 5;
  const int nst   = Cin / KSTAGE;
  const int row16 = lane & 15;
  const int hi    = lane >> 4;

  const h16* xTb = xT + (size_t)b * TP_ * Cin;
  // per-lane fragment base for the wave's two N-tiles
  const h16* Wt0 = Wp + ((size_t)(n0 >> 4) * nkc * 32 + lane) * 16;
  const h16* Wt1 = Wp + ((size_t)((n0 >> 4) + 1) * nkc * 32 + lane) * 16;

  float u = 0.0f, r = 0.0f;           // LIF carries: lane owns neuron n0+lane

  for (int t0 = 0; t0 < TP_; t0 += 16) {
    v8f c0 = {}, c1 = {};
    for (int st = 0; st < nst; ++st) {
      __syncthreads();                // ldsx consumers of previous stage done
#if __has_builtin(__builtin_amdgcn_tensor_load_to_lds)
      if (tid < 32) {                 // one wave issues the TDM descriptor
        unsigned lds_off = (unsigned)(size_t)(void*)&ldsx[0];
        unsigned long long ga =
            (unsigned long long)(size_t)(xTb + (size_t)t0 * Cin + st * KSTAGE);
        // D# group0: count=1 | lds_addr | global_addr[56:0] | type=2
        u32x4 g0 = {1u, lds_off, (unsigned)ga,
                    (unsigned)(ga >> 32) | 0x80000000u};
        // D# group1: data_size=2B, pad every 512B by 16B, 2D tile 256x16
        const unsigned TD1 = 1u << 20;          // tensor_dim1 (never OOB)
        i32x8 g1;
        g1[0] = (int)((1u << 16) | (1u << 20) | (6u << 22) | (3u << 25));
        g1[1] = (int)(((unsigned)Cin & 0xFFFFu) << 16);         // td0 lo
        g1[2] = (int)(((unsigned)Cin >> 16) | ((TD1 & 0xFFFFu) << 16));
        g1[3] = (int)((TD1 >> 16) | ((unsigned)KSTAGE << 16));  // tile_dim0
        g1[4] = 16;                                             // tile_dim1
        g1[5] = Cin;                                            // dim0 stride
        g1[6] = 0;
        g1[7] = 0;
        i32x4 gz = {0, 0, 0, 0};
#if __clang_major__ >= 23
        i32x8 gz8 = {0, 0, 0, 0, 0, 0, 0, 0};
        __builtin_amdgcn_tensor_load_to_lds(g0, g1, gz, gz, gz8, 0);
#else
        __builtin_amdgcn_tensor_load_to_lds(g0, g1, gz, gz, 0);
#endif
        __builtin_amdgcn_s_wait_tensorcnt(0);
      }
#else
      // Fallback: cooperative staging (coalesced b128 per thread, 4 iters)
      {
        const int kbase = st * KSTAGE;
#pragma unroll
        for (int it = 0; it < 4; ++it) {
          int s  = tid + it * 128;
          int rr = s >> 5;
          int cc = s & 31;
          v8h v = *(const v8h*)&xTb[(size_t)(t0 + rr) * Cin + kbase + cc * 8];
          *(v8h*)&ldsx[rr * LROW + cc * 8] = v;
        }
      }
#endif
      __syncthreads();

      const int kb = st * KKS;
      if (st + 1 < nst)
        __builtin_prefetch(Wt0 + (size_t)(kb + KKS) * 512, 0, 3);
      const h16* ax0 = &ldsx[row16 * LROW + hi * 16];
      // software pipeline: preload k-step 0, rotate
      v16h bf0 = *(const v16h*)(Wt0 + (size_t)kb * 512);
      v16h bf1 = *(const v16h*)(Wt1 + (size_t)kb * 512);
      v8h  alo = *(const v8h*)ax0;
      v8h  ahi = *(const v8h*)(ax0 + 8);
#pragma unroll
      for (int ks = 0; ks < KKS; ++ks) {
        v16h nbf0, nbf1;
        v8h  nalo, nahi;
        if (ks + 1 < KKS) {           // issue next loads before this WMMA
          nbf0 = *(const v16h*)(Wt0 + (size_t)(kb + ks + 1) * 512);
          nbf1 = *(const v16h*)(Wt1 + (size_t)(kb + ks + 1) * 512);
          nalo = *(const v8h*)(ax0 + (ks + 1) * 32);
          nahi = *(const v8h*)(ax0 + (ks + 1) * 32 + 8);
        }
        v16h af = __builtin_shufflevector(alo, ahi, 0, 1, 2, 3, 4, 5, 6, 7, 8,
                                          9, 10, 11, 12, 13, 14, 15);
        c0 = __builtin_amdgcn_wmma_f32_16x16x32_f16(false, af, false, bf0,
                                                    (short)0, c0, false, false);
        c1 = __builtin_amdgcn_wmma_f32_16x16x32_f16(false, af, false, bf1,
                                                    (short)0, c1, false, false);
        if (ks + 1 < KKS) {
          bf0 = nbf0; bf1 = nbf1; alo = nalo; ahi = nahi;
        }
      }
    }
    // Spill C tiles: lane -> Z[t0 + v + 8*hi, n] for n = (lane&15) and +16
#pragma unroll
    for (int v = 0; v < 8; ++v) {
      zbuf[wave][v + hi * 8][row16]      = c0[v];
      zbuf[wave][v + hi * 8][16 + row16] = c1[v];
    }
    // LIF scan: every lane owns neuron n0+lane; 16 sequential timesteps.
    {
      float sv[16];
#pragma unroll
      for (int j = 0; j < 16; ++j) {
        u = ALPHA_SR_ * u + zbuf[wave][j][lane];
        float m = u + r;
        float s = (m >= THETA_) ? 1.0f : 0.0f;
        r = ALPHA_REF_ * r - REF_SCALE_ * s;
        sv[j] = s;
      }
      if constexpr (F16OUT) {
#pragma unroll
        for (int j = 0; j < 16; ++j) sbuf[wave][j][lane] = (h16)sv[j];
        // cooperative coalesced store: 16 rows x 32 f16 -> [B,TP,Cout]
        int rr = lane >> 1, ch = (lane & 1) * 16;
        v8h a0, a1;
#pragma unroll
        for (int e = 0; e < 8; ++e) {
          a0[e] = sbuf[wave][rr][ch + e];
          a1[e] = sbuf[wave][rr][ch + 8 + e];
        }
        h16* dst = &sOut[((size_t)b * TP_ + t0 + rr) * Cout + n0 + ch];
        *(v8h*)dst       = a0;
        *(v8h*)(dst + 8) = a1;
      } else {
        float* yo = yOut + ((size_t)b * Cout + n0 + lane) * T_ + t0;
        if (t0 + 16 <= T_) {
#pragma unroll
          for (int j = 0; j < 16; j += 4) {
            v4f t = {sv[j], sv[j + 1], sv[j + 2], sv[j + 3]};
            *(v4f*)(yo + j) = t;
          }
        } else {
          for (int j = 0; j < 16; ++j)
            if (t0 + j < T_) yo[j] = sv[j];
        }
      }
    }
  }
}

// ---------------------------------------------------------------------------
extern "C" void kernel_launch(void* const* d_in, const int* in_sizes, int n_in,
                              void* d_out, int out_size, void* d_ws, size_t ws_size,
                              hipStream_t stream) {
  const float* x  = (const float*)d_in[0];   // [B, IN, T]
  const float* W1 = (const float*)d_in[1];   // [HID, IN]
  const float* W2 = (const float*)d_in[2];   // [OUT, HID]
  float* out = (float*)d_out;                // [B, OUT, T]

  char* ws = (char*)d_ws;
  size_t off = 0;
  h16* xT  = (h16*)(ws + off); off += (size_t)B_ * TP_ * IN_  * sizeof(h16);
  h16* Wp1 = (h16*)(ws + off); off += (size_t)HID_ * IN_      * sizeof(h16);
  h16* Wp2 = (h16*)(ws + off); off += (size_t)OUT_ * HID_     * sizeof(h16);
  h16* s1  = (h16*)(ws + off); off += (size_t)B_ * TP_ * HID_ * sizeof(h16);
  (void)ws_size; (void)in_sizes; (void)n_in; (void)out_size;

  transpose_x_kernel<<<dim3((TP_ + 31) / 32, IN_ / 32, B_), dim3(32, 8), 0,
                       stream>>>(x, xT);
  pack_w_kernel<<<(HID_ * IN_ + 255) / 256, 256, 0, stream>>>(W1, Wp1, HID_, IN_);
  pack_w_kernel<<<(OUT_ * HID_ + 255) / 256, 256, 0, stream>>>(W2, Wp2, OUT_, HID_);
  slayer_layer_kernel<true><<<dim3(B_, HID_ / 128), 128, 0, stream>>>(
      xT, Wp1, s1, nullptr, IN_, HID_);
  slayer_layer_kernel<false><<<dim3(B_, OUT_ / 128), 128, 0, stream>>>(
      s1, Wp2, nullptr, out, HID_, OUT_);
}
